// ActorDropout_33698313404512
// MI455X (gfx1250) — matrix operand order repaired and checked by reading, not compiled, not run on hardware
//
#include <hip/hip_runtime.h>

typedef __attribute__((ext_vector_type(16))) _Float16 v16h;
typedef __attribute__((ext_vector_type(8)))  float    v8f;

namespace {
constexpr int kS  = 512;   // state dim
constexpr int kH1 = 400;
constexpr int kH2 = 300;
constexpr int kA  = 64;

// tile counts (K tiles of 32, N tiles of 16), padded where needed
constexpr int KT1 = 16, NT1 = 25;   // L1: K=512,  N=400
constexpr int KT2 = 13, NT2 = 19;   // L2: K=416p, N=304p
constexpr int KT3 = 10, NT3 = 4;    // L3: K=320p, N=64

// LDS pitches (f16 elements unless noted) — chosen so dword stride mod 64
// is coprime-ish with 16 => conflict-free A-fragment reads across M lanes.
constexpr int PS = 520;  // staged state (512 cols used)
constexpr int P1 = 424;  // x1 (416 cols used; [400,416) zero pad)
constexpr int P2 = 328;  // x2 (320 cols used; [304,320) zero pad)
constexpr int PY = 66;   // y pitch in f32 (64 cols used)

constexpr int ROWS    = 128;  // batch rows per block
constexpr int THREADS = 256;  // 8 waves

constexpr int SZ_A       = ROWS * PS * 2;   // 133120 B : state, later x2
constexpr int SZ_B       = ROWS * P1 * 2;   // 108544 B : x1, later y (f32)
constexpr int SMEM_BYTES = SZ_A + SZ_B;     // 241664 B  (< 320 KB/WGP)
} // namespace

union AFrag {
  v16h v;
  unsigned u[8];
};

// ---------------------------------------------------------------------------
// Pack a row-major f32 weight W[N][K] into f16 WMMA B-operand fragments:
//   dst[((nt*KT + kt)*32 + lane)*16 + slot]
// B tile is 32(K) x 16(N): lanes 0-15 hold col n=lane, K slots {0..7,16..23};
// lanes 16-31 hold col n=lane-16, K slots {8..15,24..31}. Pads with zeros.
// ---------------------------------------------------------------------------
__global__ void pack_w(const float* __restrict__ W, _Float16* __restrict__ dst,
                       int N, int K, int KT, int total) {
  int i = blockIdx.x * blockDim.x + threadIdx.x;
  if (i >= total) return;
  int slot = i & 15;
  int lane = (i >> 4) & 31;
  int t    = i >> 9;
  int kt   = t % KT;
  int nt   = t / KT;
  int n = nt * 16 + (lane & 15);
  int k = kt * 32 + ((slot >> 3) << 4) + ((lane >> 4) << 3) + (slot & 7);
  float v = (n < N && k < K) ? W[n * K + k] : 0.0f;
  dst[i] = (_Float16)v;
}

// ---------------------------------------------------------------------------
// One GEMM layer for a 16-row slab owned by this wave.
//   sA  : f16 LDS activations, pitch PIN (K = KT*32 cols, zero padded)
//   sOut: LDS output, pitch POUT (f16 or f32)
//   pw  : packed f16 weights (B fragments), bias: f32 global
// A fragment (16x32 f16): lane<16 holds row m=lane, K slots {0..7,16..23};
// lane>=16 holds row m=lane-16, K slots {8..15,24..31}.
// ---------------------------------------------------------------------------
template <int KT, int NT, int PIN, int POUT, bool RELU, bool OUTF32>
__device__ __forceinline__ void layer(const _Float16* __restrict__ sA,
                                      void* __restrict__ sOut,
                                      const _Float16* __restrict__ pw,
                                      const float* __restrict__ bias,
                                      int Nout, int lane, int rowbase) {
  const int half  = lane >> 4;
  const int m     = lane & 15;
  const int nlane = lane & 15;

  // Load all A fragments for this slab into registers (KT * 8 VGPRs).
  AFrag a[KT];
  const _Float16* rowp = sA + (rowbase + m) * PIN;
#pragma unroll
  for (int k = 0; k < KT; ++k) {
    const int kb = k * 32 + (half ? 8 : 0);
    const unsigned* p0 = (const unsigned*)(rowp + kb);        // K = kb..kb+7
    const unsigned* p1 = (const unsigned*)(rowp + kb + 16);   // K = kb+16..kb+23
#pragma unroll
    for (int j = 0; j < 4; ++j) {
      a[k].u[j]     = p0[j];
      a[k].u[4 + j] = p1[j];
    }
  }

  for (int nt = 0; nt < NT; ++nt) {
    v8f c0 = {};
    v8f c1 = {};
    const _Float16* bw = pw + ((long)nt * KT) * 512 + lane * 16;
#pragma unroll
    for (int k = 0; k < KT; k += 2) {
      v16h b0 = *(const v16h*)(bw + k * 512);
      c0 = __builtin_amdgcn_wmma_f32_16x16x32_f16(false, a[k].v, false, b0,
                                                  (short)0, c0, false, false);
      if (k + 1 < KT) {
        v16h b1 = *(const v16h*)(bw + (k + 1) * 512);
        c1 = __builtin_amdgcn_wmma_f32_16x16x32_f16(false, a[k + 1].v, false, b1,
                                                    (short)0, c1, false, false);
      }
    }
    const int gn = nt * 16 + nlane;
    const float bv = (gn < Nout) ? bias[gn] : 0.0f;
#pragma unroll
    for (int r = 0; r < 8; ++r) {
      float val = c0[r] + c1[r] + bv;
      if constexpr (RELU) val = fmaxf(val, 0.0f);
      const int row = rowbase + half * 8 + r;
      if constexpr (OUTF32)
        ((float*)sOut)[row * POUT + gn] = val;
      else
        ((_Float16*)sOut)[row * POUT + gn] = (_Float16)val;
    }
  }
}

// ---------------------------------------------------------------------------
// Fused 3-layer MLP + positive-softmax-tanh head. One block = 128 batch rows.
// ---------------------------------------------------------------------------
__global__ __launch_bounds__(THREADS, 1)
void actor_fused(const float* __restrict__ state,
                 const float* __restrict__ b1, const float* __restrict__ b2,
                 const float* __restrict__ b3,
                 const _Float16* __restrict__ pw1,
                 const _Float16* __restrict__ pw2,
                 const _Float16* __restrict__ pw3,
                 float* __restrict__ out) {
  extern __shared__ char smem[];
  _Float16* sIn = (_Float16*)(smem);          // state (PS) -> later x2 (P2)
  _Float16* sX1 = (_Float16*)(smem + SZ_A);   // x1 (P1)    -> later y (f32, PY)
  float*    sY  = (float*)(smem + SZ_A);

  const int tid     = threadIdx.x;
  const int lane    = tid & 31;
  const int wave    = tid >> 5;
  const int rowbase = wave * 16;
  const long blockRow = (long)blockIdx.x * ROWS;

  // --- stage state tile -> LDS f16 (coalesced float4 reads) ---
  {
    const float4* gs = (const float4*)(state + blockRow * kS);
    for (int i = tid; i < ROWS * kS / 4; i += THREADS) {
      const int r  = i >> 7;       // 128 float4 per row
      const int c4 = i & 127;
      float4 v = gs[i];
      _Float16* d = sIn + r * PS + c4 * 4;
      d[0] = (_Float16)v.x; d[1] = (_Float16)v.y;
      d[2] = (_Float16)v.z; d[3] = (_Float16)v.w;
    }
    // zero x1 K-pad cols [400,416)
    for (int i = tid; i < ROWS * 16; i += THREADS)
      sX1[(i >> 4) * P1 + 400 + (i & 15)] = (_Float16)0.0f;
  }
  __syncthreads();

  // --- layer 1: state(LDS) x W1^T -> relu -> x1(LDS) ---
  layer<KT1, NT1, PS, P1, true, false>(sIn, sX1, pw1, b1, kH1, lane, rowbase);
  __syncthreads();

  // zero x2 K-pad cols [304,320) (x2 aliases the state region)
  for (int i = tid; i < ROWS * 16; i += THREADS)
    sIn[(i >> 4) * P2 + 304 + (i & 15)] = (_Float16)0.0f;
  __syncthreads();

  // --- layer 2: x1 x W2^T -> relu -> x2 (region A) ---
  layer<KT2, NT2, P1, P2, true, false>(sX1, sIn, pw2, b2, kH2, lane, rowbase);
  __syncthreads();

  // --- layer 3: x2 x W3^T + b3 -> y (f32, region B) ---
  layer<KT3, NT3, P2, PY, false, true>(sIn, sY, pw3, b3, kA, lane, rowbase);
  __syncthreads();

  // --- head: tanh on cols 0..62, positive-sum normalize; sigmoid on col 63 ---
  if (tid < ROWS) {
    const float* yr = sY + tid * PY;
    float s = 0.0f;
    for (int j = 0; j < kA - 1; ++j) {
      float t = tanhf(yr[j]);
      s += (t > 0.0f) ? t : 0.0f;
    }
    const float inv = 1.0f / s;
    float* o = out + (blockRow + tid) * kA;
    for (int j = 0; j < kA - 1; ++j) {
      float t = tanhf(yr[j]);
      o[j] = (t > 0.0f) ? t * inv : t;
    }
    o[kA - 1] = 1.0f / (1.0f + expf(-yr[kA - 1]));
  }
}

// ---------------------------------------------------------------------------
extern "C" void kernel_launch(void* const* d_in, const int* in_sizes, int n_in,
                              void* d_out, int out_size, void* d_ws,
                              size_t ws_size, hipStream_t stream) {
  const float* state = (const float*)d_in[0];
  const float* W1 = (const float*)d_in[1];
  const float* b1 = (const float*)d_in[2];
  const float* W2 = (const float*)d_in[3];
  const float* b2 = (const float*)d_in[4];
  const float* W3 = (const float*)d_in[5];
  const float* b3 = (const float*)d_in[6];
  float* out = (float*)d_out;

  constexpr int T1 = NT1 * KT1 * 512;  // 204800 f16
  constexpr int T2 = NT2 * KT2 * 512;  // 126464 f16
  constexpr int T3 = NT3 * KT3 * 512;  //  20480 f16
  _Float16* pw1 = (_Float16*)d_ws;
  _Float16* pw2 = pw1 + T1;
  _Float16* pw3 = pw2 + T2;

  pack_w<<<(T1 + 255) / 256, 256, 0, stream>>>(W1, pw1, kH1, kS,  KT1, T1);
  pack_w<<<(T2 + 255) / 256, 256, 0, stream>>>(W2, pw2, kH2, kH1, KT2, T2);
  pack_w<<<(T3 + 255) / 256, 256, 0, stream>>>(W3, pw3, kA,  kH2, KT3, T3);

  const int nBlocks = 131072 / ROWS;  // 1024
  actor_fused<<<nBlocks, THREADS, SMEM_BYTES, stream>>>(state, b1, b2, b3, pw1,
                                                        pw2, pw3, out);
}
